// MHA_80556406603947
// MI455X (gfx1250) — matrix operand order repaired and checked by reading, not compiled, not run on hardware
//
#include <hip/hip_runtime.h>
#include <hip/hip_bf16.h>

// ---------------------------------------------------------------------------
// CDNA5 (gfx1250) fused synthesizer-attention pipeline.
// bf16 WMMA with fragment-ready swizzled layouts; fp32->bf16 conversion fused
// into GEMM staging; double-buffered GEMM; batch-looped attention so
// random_mat is read from HBM once; float4-vectorized softmax passes;
// bank-conflict-free padded score tile.
// ---------------------------------------------------------------------------

typedef __attribute__((ext_vector_type(16))) __bf16 bf16x16;
typedef __attribute__((ext_vector_type(8)))  float  f32x8;

static constexpr int Bc = 4, Sc = 1024, Dc = 1024, Hc = 16, HDc = 64;
static constexpr int SROW = Sc + 4;  // padded score row stride (bank-conflict-free)

__device__ __forceinline__ f32x8 wmma_bf16(bf16x16 a, bf16x16 b, f32x8 c) {
  return __builtin_amdgcn_wmma_f32_16x16x32_bf16(
      /*neg_a=*/false, a, /*neg_b=*/false, b,
      /*c_mod=*/(short)0, c, /*reuse_a=*/false, /*reuse_b=*/false);
}

__device__ __forceinline__ unsigned pack_bf16(float x, float y) {
  union { __bf16 h[2]; unsigned u; } pk;
  pk.h[0] = (__bf16)x;
  pk.h[1] = (__bf16)y;
  return pk.u;
}

__device__ __forceinline__ float wave_reduce_max(float v) {
#pragma unroll
  for (int m = 16; m > 0; m >>= 1) v = fmaxf(v, __shfl_xor(v, m, 32));
  return v;
}
__device__ __forceinline__ float wave_reduce_sum(float v) {
#pragma unroll
  for (int m = 16; m > 0; m >>= 1) v += __shfl_xor(v, m, 32);
  return v;
}

// A-fragment (16x32): lane = (half<<4)|row; elem(k) = 2*i2 + (k&1),
//   i2 = ((k>>4)<<2)|((k&7)>>1), half = (k>>3)&1.
// B-fragment (32x16): lane = (((k>>3)&1)<<4)|(n&15); elem = ((k>>4)<<3)|(k&7).

// ---------------- weight swizzle: [K,N] fp32 -> B-fragment bf16 ------------
__global__ void w_swizzle_b_kernel(const float* __restrict__ src,
                                   __bf16* __restrict__ dst, int K, int N) {
  int idx = blockIdx.x * blockDim.x + threadIdx.x;
  if (idx >= K * N) return;
  int k = idx / N, n = idx - k * N;
  int kt = k >> 5, kk = k & 31;
  int j = n >> 4;
  int ln = (((kk >> 3) & 1) << 4) | (n & 15);
  int m = ((kk >> 4) << 3) | (kk & 7);
  dst[(((size_t)j * (K >> 5) + kt) * 32 + ln) * 16 + m] = (__bf16)src[idx];
}

// ---------------- double-buffered GEMM, conversion fused into staging ------
// OUT_MODE 1: fp32 row-major [M,N]
// OUT_MODE 2: bf16 Q A-fragment-swizzled [B,H][st][kt][32][16]
// OUT_MODE 3: bf16 V B-fragment-swizzled [B,H][kt][j][32][16]
template <int OUT_MODE, bool AF32, bool BF32>
__global__ __launch_bounds__(128) void gemm_kernel(
    const void* __restrict__ Ap, const void* __restrict__ Bp,
    const float* __restrict__ bias, void* __restrict__ Cout,
    int M, int N, int K) {
  __shared__ __bf16 sAf[2][2048];  // [buf][wave-tile][lane][elem]
  __shared__ __bf16 sBf[2][2048];  // [buf][j-tile][lane][m]

  const int t = threadIdx.x, lane = t & 31, w = t >> 5;
  const int col0 = blockIdx.x * 64, row0 = blockIdx.y * 64;
  // A staging: thread covers (row0+ar, k0+ac0 .. +15)
  const int ar = t >> 1, ac0 = (t & 1) << 4, awt = ar >> 4, arow = ar & 15;
  // B staging: thread covers rows (k0+br, k0+br+1), cols col0+bc0..+7
  const int br = (t >> 3) << 1, bc0 = (t & 7) << 3;

  unsigned aU[8];  // staged A dwords (2 packed bf16, consecutive k)
  unsigned bU[8];  // staged B dwords (2 packed bf16, row pair br/br+1)
  f32x8 acc[4] = {};

  auto globalLoad = [&](int k0) {
    if constexpr (AF32) {
      const float* ga = (const float*)Ap + (size_t)(row0 + ar) * K + k0 + ac0;
      float v[16];
      *(float4*)&v[0]  = *(const float4*)(ga + 0);
      *(float4*)&v[4]  = *(const float4*)(ga + 4);
      *(float4*)&v[8]  = *(const float4*)(ga + 8);
      *(float4*)&v[12] = *(const float4*)(ga + 12);
#pragma unroll
      for (int e = 0; e < 8; ++e) aU[e] = pack_bf16(v[2 * e], v[2 * e + 1]);
    } else {
      const __bf16* ga = (const __bf16*)Ap + (size_t)(row0 + ar) * K + k0 + ac0;
      *(uint4*)&aU[0] = *(const uint4*)ga;
      *(uint4*)&aU[4] = *(const uint4*)(ga + 8);
    }
    if constexpr (BF32) {
      const float* gb = (const float*)Bp + (size_t)(k0 + br) * N + col0 + bc0;
      float v0[8], v1[8];
      *(float4*)&v0[0] = *(const float4*)(gb + 0);
      *(float4*)&v0[4] = *(const float4*)(gb + 4);
      *(float4*)&v1[0] = *(const float4*)(gb + N);
      *(float4*)&v1[4] = *(const float4*)(gb + N + 4);
#pragma unroll
      for (int e = 0; e < 8; ++e) bU[e] = pack_bf16(v0[e], v1[e]);
    } else {
      const __bf16* gb = (const __bf16*)Bp + (size_t)(k0 + br) * N + col0 + bc0;
      __bf16 t0[8], t1[8];
      *(uint4*)&t0[0] = *(const uint4*)gb;
      *(uint4*)&t1[0] = *(const uint4*)(gb + N);
#pragma unroll
      for (int e = 0; e < 8; ++e) {
        union { __bf16 h[2]; unsigned u; } pk;
        pk.h[0] = t0[e];
        pk.h[1] = t1[e];
        bU[e] = pk.u;
      }
    }
  };

  auto ldsStore = [&](int buf) {
#pragma unroll
    for (int e = 0; e < 8; ++e) {
      int k = ac0 + 2 * e;  // even
      int hl = (k >> 3) & 1;
      int i2 = ((k >> 4) << 2) | ((k & 7) >> 1);
      int ln = (hl << 4) | arow;
      *(unsigned*)&sAf[buf][((awt * 32 + ln) << 4) + 2 * i2] = aU[e];
    }
    int hlb = (br >> 3) & 1, m = ((br >> 4) << 3) | (br & 7);  // m even
#pragma unroll
    for (int e = 0; e < 8; ++e) {
      int n = bc0 + e, j = n >> 4;
      int ln = (hlb << 4) | (n & 15);
      *(unsigned*)&sBf[buf][((j * 32 + ln) << 4) + m] = bU[e];
    }
  };

  auto compute = [&](int buf) {
    bf16x16 aF = *(const bf16x16*)&sAf[buf][(w * 32 + lane) << 4];
#pragma unroll
    for (int j = 0; j < 4; ++j) {
      bf16x16 bF = *(const bf16x16*)&sBf[buf][(j * 32 + lane) << 4];
      acc[j] = wmma_bf16(aF, bF, acc[j]);
    }
  };

  const int nk = K >> 5;
  globalLoad(0);
  ldsStore(0);
  __syncthreads();
  for (int kt = 0; kt < nk; ++kt) {
    if (kt + 1 < nk) globalLoad((kt + 1) << 5);  // loads fly under the WMMAs
    compute(kt & 1);
    if (kt + 1 < nk) ldsStore((kt + 1) & 1);     // writes the other buffer
    __syncthreads();
  }

  const int half = lane >> 4;
#pragma unroll
  for (int j = 0; j < 4; ++j) {
    int gcol = col0 + 16 * j + (lane & 15);
    float bv = bias[gcol];
#pragma unroll
    for (int i = 0; i < 8; ++i) {
      int grow = row0 + 16 * w + 8 * half + i;
      float v = acc[j][i] + bv;
      if constexpr (OUT_MODE == 1) {
        ((float*)Cout)[(size_t)grow * N + gcol] = v;
      } else if constexpr (OUT_MODE == 2) {  // Q -> A-fragment-swizzled
        int bb = grow >> 10, ss = grow & (Sc - 1);
        int hh = gcol >> 6, dd = gcol & (HDc - 1);
        int st = ss >> 4, rr = ss & 15;
        int kt = dd >> 5, kk = dd & 31;
        int hlq = (kk >> 3) & 1;
        int i2 = ((kk >> 4) << 2) | ((kk & 7) >> 1);
        int el = 2 * i2 + (kk & 1);
        int ln = (hlq << 4) | rr;
        ((__bf16*)Cout)[((((size_t)(bb * Hc + hh) * 64 + st) * 2 + kt) * 32 + ln) * 16 + el] =
            (__bf16)v;
      } else {  // OUT_MODE 3: V -> B-fragment-swizzled
        int bb = grow >> 10, ss = grow & (Sc - 1);
        int hh = gcol >> 6, dd = gcol & (HDc - 1);
        int kt = ss >> 5, kk = ss & 31;
        int jj = dd >> 4;
        int ln = (((kk >> 3) & 1) << 4) | (dd & 15);
        int m = ((kk >> 4) << 3) | (kk & 7);
        ((__bf16*)Cout)[((((size_t)(bb * Hc + hh) * 32 + kt) * 4 + jj) * 32 + ln) * 16 + m] =
            (__bf16)v;
      }
    }
  }
}

// ---------------- fused synthesizer attention ------------------------------
// One block = (h, 16-row q-tile); loops over batch so each random_mat row is
// fetched from HBM once and its softmax stats are reused for all 4 batches.
__global__ __launch_bounds__(128) void attn_fused_kernel(
    const __bf16* __restrict__ Qsw,   // [B,H][64 st][2 kt][32][16]
    const __bf16* __restrict__ Vsw,   // [B,H][32 kt][4 j][32][16]
    const __bf16* __restrict__ W1sw,  // [4 j][2 kt][32][16]
    const float* __restrict__ b1,
    const __bf16* __restrict__ W2sw,  // [64 j][2 kt][32][16]
    const float* __restrict__ b2,
    const float* __restrict__ rand_mat,  // [H,S,S] fp32
    const float* __restrict__ a1p, const float* __restrict__ a2p,
    __bf16* __restrict__ AttnOut) {      // [B,S,H,HD] bf16 row-major
  // LDS carve (SROW-padded fp32 score tile is bank-conflict-free)
  constexpr int SCORE_B = 16 * SROW * 4;  // 65792
  extern __shared__ char smem[];
  float*  sScore    = (float*)smem;
  __bf16* sPf       = (__bf16*)(smem + SCORE_B);                  // 32768 B
  __bf16* sHidF     = (__bf16*)(smem + SCORE_B + 32768);          //  2048 B
  float*  sRmax     = (float*)(smem + SCORE_B + 32768 + 2048);    //    64 B
  float*  sRinv     = (float*)(smem + SCORE_B + 32768 + 2112);    //    64 B
  float*  sRowScale = (float*)(smem + SCORE_B + 32768 + 2176);    //    64 B

  const int t = threadIdx.x, lane = t & 31, w = t >> 5;
  const int blk = blockIdx.x;  // 0..1023
  const int qt = blk & 63;
  const int h = blk >> 6;
  const int q0 = qt * 16;
  const int half = lane >> 4;

  const float a1 = a1p[0], a2 = a2p[0];
  const float s1 = a1 / (a1 + a2), s2 = a2 / (a1 + a2);

  // ---- phase 0: rand-branch softmax stats, once per (h, q-tile) ----
  for (int r = w; r < 16; r += 4) {
    const float4* rr4 = (const float4*)(rand_mat + ((size_t)h * Sc + q0 + r) * Sc);
    float rmax = -3.402823466e38f;
    for (int c = lane; c < Sc / 4; c += 32) {
      float4 v = rr4[c];
      rmax = fmaxf(rmax, fmaxf(fmaxf(v.x, v.y), fmaxf(v.z, v.w)));
    }
    rmax = wave_reduce_max(rmax);
    float rsum = 0.0f;
    for (int c = lane; c < Sc / 4; c += 32) {
      float4 v = rr4[c];
      rsum += (__expf(v.x - rmax) + __expf(v.y - rmax)) +
              (__expf(v.z - rmax) + __expf(v.w - rmax));
    }
    rsum = wave_reduce_sum(rsum);
    if (lane == 0) {
      sRmax[r] = rmax;
      sRinv[r] = 1.0f / rsum;
    }
  }
  __syncthreads();

  // batch-invariant W1 fragments (live in VGPRs across the whole kernel)
  const bf16x16 bW0 = *(const bf16x16*)(W1sw + ((w * 2 + 0) * 32 + lane) * 16);
  const bf16x16 bW1 = *(const bf16x16*)(W1sw + ((w * 2 + 1) * 32 + lane) * 16);

  // invariant swizzle pieces for the P store (k0 = (4*lane) & 31)
  const int pk0 = (4 * lane) & 31;
  const int phl = (pk0 >> 3) & 1;
  const int pel = 2 * (((pk0 >> 4) << 2) | ((pk0 & 7) >> 1));

  for (int b = 0; b < Bc; ++b) {
    const int bh = b * Hc + h;

    // ---- hidden = relu(Q @ W1 + b1); wave w owns cols 16w..16w+15 ----
    {
      const __bf16* qbase = Qsw + (((size_t)bh * 64 + qt) * 2) * 512;
      bf16x16 aQ0 = *(const bf16x16*)(qbase + lane * 16);
      bf16x16 aQ1 = *(const bf16x16*)(qbase + 512 + lane * 16);
      f32x8 acc = {};
      acc = wmma_bf16(aQ0, bW0, acc);
      acc = wmma_bf16(aQ1, bW1, acc);
      int col = 16 * w + (lane & 15);
      float bias = b1[col];
      int kt = col >> 5, kk = col & 31;
      int hlq = (kk >> 3) & 1;
      int i2 = ((kk >> 4) << 2) | ((kk & 7) >> 1);
      int el = 2 * i2 + (kk & 1);
#pragma unroll
      for (int i = 0; i < 8; ++i) {
        int row = 8 * half + i;
        int ln = (hlq << 4) | row;
        sHidF[((kt * 32 + ln) << 4) + el] = (__bf16)fmaxf(acc[i] + bias, 0.0f);
      }
    }
    __syncthreads();

    // ---- dense scores [16,1024] = hidden @ W2 + b2 -> padded LDS fp32 ----
    {
      bf16x16 aH0 = *(const bf16x16*)&sHidF[(0 * 32 + lane) << 4];
      bf16x16 aH1 = *(const bf16x16*)&sHidF[(1 * 32 + lane) << 4];
      for (int j = w; j < 64; j += 4) {
        bf16x16 bb0 = *(const bf16x16*)(W2sw + (((size_t)j * 2 + 0) * 32 + lane) * 16);
        bf16x16 bb1 = *(const bf16x16*)(W2sw + (((size_t)j * 2 + 1) * 32 + lane) * 16);
        f32x8 acc = {};
        acc = wmma_bf16(aH0, bb0, acc);
        acc = wmma_bf16(aH1, bb1, acc);
        int col = j * 16 + (lane & 15);
        float bias = b2[col];
#pragma unroll
        for (int i = 0; i < 8; ++i) {
          int row = 8 * half + i;
          sScore[row * SROW + col] = acc[i] + bias;  // conflict-free banks
        }
      }
    }
    __syncthreads();

    // ---- per-row mix + softmax (float4-vectorized); P -> A-swizzled ----
    for (int r = w; r < 16; r += 4) {
      const float rmax = sRmax[r], rinv = sRinv[r];
      const float4* rr4 = (const float4*)(rand_mat + ((size_t)h * Sc + q0 + r) * Sc);
      float4* srow4 = (float4*)(sScore + (size_t)r * SROW);

      float cmax = -3.402823466e38f;
      for (int c = lane; c < Sc / 4; c += 32) {
        float4 rv = rr4[c];  // hot in WGP$/L2 after phase 0 / earlier batches
        float4 sv = srow4[c];
        sv.x = s1 * (__expf(rv.x - rmax) * rinv) + s2 * sv.x;
        sv.y = s1 * (__expf(rv.y - rmax) * rinv) + s2 * sv.y;
        sv.z = s1 * (__expf(rv.z - rmax) * rinv) + s2 * sv.z;
        sv.w = s1 * (__expf(rv.w - rmax) * rinv) + s2 * sv.w;
        srow4[c] = sv;
        cmax = fmaxf(cmax, fmaxf(fmaxf(sv.x, sv.y), fmaxf(sv.z, sv.w)));
      }
      cmax = wave_reduce_max(cmax);

      const int lnp = (phl << 4) | r;
      float psum = 0.0f;
      for (int c = lane; c < Sc / 4; c += 32) {
        float4 sv = srow4[c];
        float p0 = __expf(sv.x - cmax), p1 = __expf(sv.y - cmax);
        float p2 = __expf(sv.z - cmax), p3 = __expf(sv.w - cmax);
        psum += (p0 + p1) + (p2 + p3);
        int kt = c >> 3;
        union { unsigned u[2]; uint2 v; } pk;
        pk.u[0] = pack_bf16(p0, p1);
        pk.u[1] = pack_bf16(p2, p3);
        *(uint2*)&sPf[((kt * 32 + lnp) << 4) + pel] = pk.v;  // ds_store_b64
      }
      psum = wave_reduce_sum(psum);
      if (lane == 0) sRowScale[r] = 1.0f / psum;
    }
    __syncthreads();

    // ---- out = (P @ V) * rowScale; wave w owns HD cols 16w..16w+15 ----
    {
      const __bf16* vbase = Vsw + (size_t)bh * 32 * 4 * 512;
      f32x8 acc = {};
#pragma unroll 4
      for (int kt = 0; kt < 32; ++kt) {
        bf16x16 aP = *(const bf16x16*)&sPf[(kt * 32 + lane) << 4];
        bf16x16 bV = *(const bf16x16*)(vbase + (((size_t)kt * 4 + w) * 32 + lane) * 16);
        acc = wmma_bf16(aP, bV, acc);
      }
      int col = 16 * w + (lane & 15);
#pragma unroll
      for (int i = 0; i < 8; ++i) {
        int row = 8 * half + i;
        float v = acc[i] * sRowScale[row];
        AttnOut[(((size_t)b * Sc + (q0 + row)) * Hc + h) * HDc + col] = (__bf16)v;
      }
    }
    __syncthreads();  // sScore/sPf/sHidF reused by next batch
  }
}

// ---------------------------------------------------------------------------

extern "C" void kernel_launch(void* const* d_in, const int* in_sizes, int n_in,
                              void* d_out, int out_size, void* d_ws, size_t ws_size,
                              hipStream_t stream) {
  const float* query    = (const float*)d_in[0];
  const float* value    = (const float*)d_in[1];
  const float* Wq       = (const float*)d_in[2];
  const float* bq       = (const float*)d_in[3];
  const float* Wv       = (const float*)d_in[4];
  const float* bv       = (const float*)d_in[5];
  const float* W1       = (const float*)d_in[6];
  const float* b1       = (const float*)d_in[7];
  const float* W2       = (const float*)d_in[8];
  const float* b2       = (const float*)d_in[9];
  const float* rand_mat = (const float*)d_in[10];
  const float* alpha1   = (const float*)d_in[11];
  const float* alpha2   = (const float*)d_in[12];
  const float* Wo       = (const float*)d_in[13];
  const float* bo       = (const float*)d_in[14];

  char* ws = (char*)d_ws;
  size_t off = 0;
  auto alloc = [&](size_t bytes) -> void* {
    void* p = ws + off;
    off += (bytes + 255) & ~(size_t)255;
    return p;
  };

  __bf16* W1sw = (__bf16*)alloc((size_t)HDc * HDc * 2);           // B-swizzled
  __bf16* W2sw = (__bf16*)alloc((size_t)HDc * Sc * 2);            // B-swizzled
  __bf16* Qsw  = (__bf16*)alloc((size_t)Bc * Hc * Sc * HDc * 2);  // A-swizzled
  __bf16* Vsw  = (__bf16*)alloc((size_t)Bc * Hc * Sc * HDc * 2);  // B-swizzled
  __bf16* Ao   = (__bf16*)alloc((size_t)Bc * Sc * Hc * HDc * 2);  // [B,S,H,HD]

  const int thr = 256;
  w_swizzle_b_kernel<<<(HDc * HDc + thr - 1) / thr, thr, 0, stream>>>(W1, W1sw, HDc, HDc);
  w_swizzle_b_kernel<<<(HDc * Sc + thr - 1) / thr, thr, 0, stream>>>(W2, W2sw, HDc, Sc);

  const int M = Bc * Sc;       // 4096
  const int N = Hc * HDc;      // 1024
  const int K = Dc;            // 1024
  dim3 gGemm(N / 64, M / 64);  // (16, 64)

  // Projections (fp32 inputs, conversion fused into staging)
  gemm_kernel<2, true, true><<<gGemm, 128, 0, stream>>>(query, Wq, bq, Qsw, M, N, K);
  gemm_kernel<3, true, true><<<gGemm, 128, 0, stream>>>(value, Wv, bv, Vsw, M, N, K);

  // Fused synthesizer attention: grid over (h, q-tile), batch looped inside.
  constexpr size_t ATTN_SMEM = (size_t)16 * SROW * 4 + 32768 + 2048 + 3 * 64;  // ~100.8 KB
  attn_fused_kernel<<<Hc * (Sc / 16), 128, ATTN_SMEM, stream>>>(
      Qsw, Vsw, W1sw, b1, W2sw, b2, rand_mat, alpha1, alpha2, Ao);

  // Output projection: [B*S, H*HD](bf16) @ [H*HD, D](fp32->bf16) -> fp32 d_out
  gemm_kernel<1, false, true><<<gGemm, 128, 0, stream>>>(Ao, Wo, bo, d_out, M, Dc, Hc * HDc);
}
